// MultiHeadAttention_22351009808584
// MI455X (gfx1250) — matrix operand order repaired and checked by reading, not compiled, not run on hardware
//
#include <hip/hip_runtime.h>
#include <hip/hip_bf16.h>

// ---------------------------------------------------------------------------
// Causal multi-head attention forward for MI455X (gfx1250), bf16 WMMA path
// with Tensor-Data-Mover (TDM) double-buffered LDS staging of shared tiles.
//   B=2, S=2048, D=2048, H=16, DK=128
// ---------------------------------------------------------------------------

typedef __bf16 bf16_t;
typedef __attribute__((ext_vector_type(16))) __bf16 v16bf;
typedef __attribute__((ext_vector_type(8)))  __bf16 v8bf;
typedef __attribute__((ext_vector_type(8)))  float  v8f;
typedef unsigned int       u32;
typedef unsigned long long u64;
typedef __attribute__((ext_vector_type(4))) u32 v4u;
typedef __attribute__((ext_vector_type(8))) int v8i;
typedef __attribute__((ext_vector_type(4))) int v4i;

constexpr int Bc  = 2;
constexpr int Sc  = 2048;
constexpr int Dc  = 2048;
constexpr int Hc  = 16;
constexpr int DKc = 128;
constexpr int Mrows = Bc * Sc;          // 4096

#if defined(__HIP_DEVICE_COMPILE__) && \
    __has_builtin(__builtin_amdgcn_tensor_load_to_lds) && \
    __has_builtin(__builtin_amdgcn_s_wait_tensorcnt)
#define HAVE_TDM 1
#else
#define HAVE_TDM 0
#endif

static __device__ __forceinline__ v8f wmma_bf16(v16bf a, v16bf b, v8f c) {
    return __builtin_amdgcn_wmma_f32_16x16x32_bf16(false, a, false, b, (short)0, c,
                                                   false, false);
}

static __device__ __forceinline__ v16bf ld_afrag(const bf16_t* p0, const bf16_t* p1) {
    v8bf lo = *(const v8bf*)p0;
    v8bf hi = *(const v8bf*)p1;
    v16bf r;
#pragma unroll
    for (int i = 0; i < 8; ++i) { r[i] = lo[i]; r[8 + i] = hi[i]; }
    return r;
}

static __device__ __forceinline__ v8f vzero8() {
    v8f z;
#pragma unroll
    for (int i = 0; i < 8; ++i) z[i] = 0.0f;
    return z;
}

#if HAVE_TDM
// Issue one TDM 2-D tile load: tile_h rows of tile_w bf16 elements, row stride
// `stride_elems`, from global `gsrc` into LDS byte offset `lds_off`.
// D# layout per CDNA5 ISA (2-byte elements, no pad,
// no multicast, no iterate, groups 2/3 zero => 2-D tensor).
static __device__ __forceinline__ void tdm_load_2d(const void* gsrc, u32 lds_off,
                                                   u32 tile_w, u32 tile_h,
                                                   u64 stride_elems) {
    const u64 ga = (u64)gsrc;
    v4u g0;
    g0[0] = 1u;                                   // count=1: valid user descriptor
    g0[1] = lds_off;                              // lds_addr
    g0[2] = (u32)ga;                              // global_addr[31:0]
    g0[3] = (u32)(ga >> 32) | (2u << 30);         // global_addr[56:32] | type=2

    const u32 td0 = 1u << 20;                     // generous tensor dims: tiles
    const u32 td1 = 1u << 20;                     // are fully in-bounds
    v8i g1;
    g1[0] = 0x00010000;                           // data_size=1 (2-byte elements)
    g1[1] = (int)((td0 & 0xFFFFu) << 16);         // tensor_dim0[15:0]
    g1[2] = (int)(((td0 >> 16) & 0xFFFFu) | ((td1 & 0xFFFFu) << 16));
    g1[3] = (int)(((td1 >> 16) & 0xFFFFu) | ((tile_w & 0xFFFFu) << 16)); // tile_dim0
    g1[4] = (int)(tile_h & 0xFFFFu);              // tile_dim1 (tile_dim2=0 -> 2D)
    g1[5] = (int)(u32)(stride_elems & 0xFFFFFFFFull);   // tensor_dim0_stride[31:0]
    g1[6] = (int)(u32)((stride_elems >> 32) & 0xFFFFull);
    g1[7] = 0;

    v4i gz; gz[0] = 0; gz[1] = 0; gz[2] = 0; gz[3] = 0;
#if __clang_major__ >= 23
    v8i gz8; gz8[0]=0; gz8[1]=0; gz8[2]=0; gz8[3]=0; gz8[4]=0; gz8[5]=0; gz8[6]=0; gz8[7]=0;
    __builtin_amdgcn_tensor_load_to_lds(g0, g1, gz, gz, gz8, 0);
#else
    __builtin_amdgcn_tensor_load_to_lds(g0, g1, gz, gz, 0);
#endif
}
#endif  // HAVE_TDM

// ---------------------------------------------------------------------------
// fp32 -> bf16 convert (grid-stride)
// ---------------------------------------------------------------------------
__global__ void cvt_f32_bf16(const float* __restrict__ in, bf16_t* __restrict__ out,
                             int n) {
    int i = blockIdx.x * blockDim.x + threadIdx.x;
    int stride = gridDim.x * blockDim.x;
    for (; i < n; i += stride) out[i] = (bf16_t)in[i];
}

// ---------------------------------------------------------------------------
// C = A @ W^T.  A: [Mrows,Dc] bf16. W: [Dc,Dc] bf16 row-major.
// Block = 256 thr = 8 waves; block tile 128 rows x 64 cols; W tile (64x32,
// shared by all 8 waves) staged in LDS, double-buffered via TDM.
// MODE 0: bf16 row-major | MODE 1: bf16 per-head transpose (vT) | MODE 2: f32
// ---------------------------------------------------------------------------
template <int MODE>
__global__ __launch_bounds__(256) void gemm_xwt(const bf16_t* __restrict__ A,
                                                const bf16_t* __restrict__ W,
                                                bf16_t* __restrict__ outb,
                                                float* __restrict__ outf) {
    __shared__ __align__(16) bf16_t bbuf[2][64][32];

    const int tid  = threadIdx.x;
    const int lane = tid & 31;
    const int w    = tid >> 5;
    const int half = lane >> 4;
    const int lm   = lane & 15;
    const int m0   = blockIdx.y * 128 + w * 16;
    const int n0   = blockIdx.x * 64;

    v8f acc[4];
#pragma unroll
    for (int t = 0; t < 4; ++t) acc[t] = vzero8();

    const bf16_t* arow = A + (size_t)(m0 + lm) * Dc;
    constexpr int KT = Dc / 32;   // 64 k-chunks

    auto stage = [&](int bi, int kc) {
#if HAVE_TDM
        if (w == 0)
            tdm_load_2d(W + (size_t)n0 * Dc + kc * 32,
                        (u32)(u64)(const void*)&bbuf[bi][0][0], 32, 64, (u64)Dc);
#else
        const int row = tid >> 2, o = (tid & 3) * 8;   // 256 thr x 16B = 4KB tile
        *(v8bf*)&bbuf[bi][row][o] =
            *(const v8bf*)(W + (size_t)(n0 + row) * Dc + kc * 32 + o);
#endif
    };

    stage(0, 0);
    for (int kc = 0; kc < KT; ++kc) {
        const int cur = kc & 1;
        if (kc + 1 < KT) {
            stage(cur ^ 1, kc + 1);
#if HAVE_TDM
            if (w == 0) __builtin_amdgcn_s_wait_tensorcnt(1);
#endif
        } else {
#if HAVE_TDM
            if (w == 0) __builtin_amdgcn_s_wait_tensorcnt(0);
#endif
        }
        __syncthreads();

        const bf16_t* ap = arow + kc * 32 + half * 16;
        v16bf a = ld_afrag(ap, ap + 8);
#pragma unroll
        for (int t = 0; t < 4; ++t) {
            v16bf bfr = *(const v16bf*)&bbuf[cur][t * 16 + lm][half * 16];
            acc[t] = wmma_bf16(a, bfr, acc[t]);
        }
        __syncthreads();
    }

#pragma unroll
    for (int t = 0; t < 4; ++t) {
#pragma unroll
        for (int r = 0; r < 8; ++r) {
            const int gm = m0 + r + 8 * half;     // C layout: M = r + 8*(lane/16)
            const int gn = n0 + t * 16 + lm;      // N = lane%16
            const float v = acc[t][r];
            if (MODE == 2) {
                outf[(size_t)gm * Dc + gn] = v;
            } else if (MODE == 1) {
                const int bb = gm >> 11;          // / Sc
                const int ss = gm & (Sc - 1);
                const int hh = gn >> 7;           // / DKc
                const int dk = gn & (DKc - 1);
                outb[(((size_t)(bb * Hc + hh)) * DKc + dk) * Sc + ss] = (bf16_t)v;
            } else {
                outb[(size_t)gm * Dc + gn] = (bf16_t)v;
            }
        }
    }
}

// ---------------------------------------------------------------------------
// Flash attention. One block = 8 waves = 128 query rows of one (b,h).
// K tile (32x128) and V tile (128x32, from vT) are shared by all 8 waves:
// staged in LDS, double-buffered via TDM. Uniform KV trip count per block.
// ---------------------------------------------------------------------------
__global__ __launch_bounds__(256) void flash_attn(const bf16_t* __restrict__ Q,
                                                  const bf16_t* __restrict__ K,
                                                  const bf16_t* __restrict__ VT,
                                                  bf16_t* __restrict__ O) {
    __shared__ __align__(16) bf16_t kbuf[2][32][128];  // [kv][dk]
    __shared__ __align__(16) bf16_t vbuf[2][128][32];  // [dk][kv]
    __shared__ __align__(16) bf16_t plds[8][16][32];   // per-wave P staging

    const int tid  = threadIdx.x;
    const int lane = tid & 31;
    const int w    = tid >> 5;
    const int half = lane >> 4;
    const int lm   = lane & 15;

    const int tilesPerBH = Sc / 128;
    const int bh   = blockIdx.x / tilesPerBH;
    const int tile = blockIdx.x % tilesPerBH;
    const int b    = bh / Hc;
    const int h    = bh % Hc;
    const int q0   = tile * 128 + w * 16;

    const size_t kbase = (size_t)(b * Sc) * Dc + h * DKc;      // + kv*Dc
    const size_t vbase = ((size_t)(b * Hc + h)) * DKc * Sc;    // + dk*Sc + kv

    auto stage = [&](int bi, int kv0) {
#if HAVE_TDM
        if (w == 0) {
            tdm_load_2d(K + kbase + (size_t)kv0 * Dc,
                        (u32)(u64)(const void*)&kbuf[bi][0][0], 128, 32, (u64)Dc);
            tdm_load_2d(VT + vbase + kv0,
                        (u32)(u64)(const void*)&vbuf[bi][0][0], 32, 128, (u64)Sc);
        }
#else
        for (int c = tid; c < 512; c += 256) {                 // 32 rows x 256B
            const int row = c >> 4, o = (c & 15) * 8;
            *(v8bf*)&kbuf[bi][row][o] =
                *(const v8bf*)(K + kbase + (size_t)(kv0 + row) * Dc + o);
        }
        for (int c = tid; c < 512; c += 256) {                 // 128 rows x 64B
            const int row = c >> 2, o = (c & 3) * 8;
            *(v8bf*)&vbuf[bi][row][o] =
                *(const v8bf*)(VT + vbase + (size_t)row * Sc + kv0 + o);
        }
#endif
    };

    // Q strip fragments: 16 rows x 128 (4 chunks of K=32)
    v16bf aq[4];
    const bf16_t* qrow = Q + (size_t)(b * Sc + q0 + lm) * Dc + h * DKc;
#pragma unroll
    for (int c = 0; c < 4; ++c) {
        const bf16_t* p = qrow + c * 32 + half * 16;
        aq[c] = ld_afrag(p, p + 8);
    }

    v8f o[8];
#pragma unroll
    for (int t = 0; t < 8; ++t) o[t] = vzero8();
    float rmax[8], rsum[8];
#pragma unroll
    for (int r = 0; r < 8; ++r) { rmax[r] = -3.0e38f; rsum[r] = 0.0f; }

    const float scale = 0.08838834764831845f;   // 1/sqrt(128)
    const int ntiles = (tile * 128 + 128) / 32; // block-uniform causal frontier

    stage(0, 0);
    for (int it = 0; it < ntiles; ++it) {
        const int kv0 = it * 32;
        const int cur = it & 1;
        if (it + 1 < ntiles) {
            stage(cur ^ 1, kv0 + 32);
#if HAVE_TDM
            if (w == 0) __builtin_amdgcn_s_wait_tensorcnt(2);
#endif
        } else {
#if HAVE_TDM
            if (w == 0) __builtin_amdgcn_s_wait_tensorcnt(0);
#endif
        }
        __syncthreads();

        // ---- scores: two 16x16 tiles from LDS K tile ----
        v8f sc[2];
#pragma unroll
        for (int j = 0; j < 2; ++j) {
            v8f s = vzero8();
#pragma unroll
            for (int c = 0; c < 4; ++c) {
                v16bf bk = *(const v16bf*)&kbuf[cur][j * 16 + lm][c * 32 + half * 16];
                s = wmma_bf16(aq[c], bk, s);
            }
            sc[j] = s;
        }

        // ---- online softmax + stage P into LDS ----
#pragma unroll
        for (int r = 0; r < 8; ++r) {
            const int qi = q0 + r + 8 * half;
            float s0 = sc[0][r] * scale;
            float s1 = sc[1][r] * scale;
            if (kv0 + lm      > qi) s0 = -1.0e30f;   // causal mask
            if (kv0 + 16 + lm > qi) s1 = -1.0e30f;

            float m = fmaxf(s0, s1);
            m = fmaxf(m, __shfl_xor(m, 1, 32));
            m = fmaxf(m, __shfl_xor(m, 2, 32));
            m = fmaxf(m, __shfl_xor(m, 4, 32));
            m = fmaxf(m, __shfl_xor(m, 8, 32));

            const float nm   = fmaxf(rmax[r], m);
            const float corr = __expf(rmax[r] - nm);
            rmax[r] = nm;

            const float p0 = __expf(s0 - nm);
            const float p1 = __expf(s1 - nm);
            float ps = p0 + p1;
            ps += __shfl_xor(ps, 1, 32);
            ps += __shfl_xor(ps, 2, 32);
            ps += __shfl_xor(ps, 4, 32);
            ps += __shfl_xor(ps, 8, 32);
            rsum[r] = rsum[r] * corr + ps;

#pragma unroll
            for (int t = 0; t < 8; ++t) o[t][r] *= corr;

            const int mrow = r + 8 * half;
            plds[w][mrow][lm]      = (bf16_t)p0;
            plds[w][mrow][16 + lm] = (bf16_t)p1;
        }
        __syncthreads();

        // ---- P @ V from LDS V tile ----
        const bf16_t* pl = &plds[w][lm][half * 16];
        v16bf ap = ld_afrag(pl, pl + 8);
#pragma unroll
        for (int t = 0; t < 8; ++t) {
            v16bf bv = *(const v16bf*)&vbuf[cur][t * 16 + lm][half * 16];
            o[t] = wmma_bf16(ap, bv, o[t]);
        }
        __syncthreads();   // protects buffer/plds reuse before next stage()
    }

    // ---- normalize and write attention output (bf16 [B,S,D]) ----
#pragma unroll
    for (int r = 0; r < 8; ++r) {
        const float inv = 1.0f / rsum[r];
        const int s = q0 + r + 8 * half;
        bf16_t* orow = O + (size_t)(b * Sc + s) * Dc + h * DKc;
#pragma unroll
        for (int t = 0; t < 8; ++t) orow[t * 16 + lm] = (bf16_t)(o[t][r] * inv);
    }
}

// ---------------------------------------------------------------------------
// Host-side launch. Workspace layout (bytes):
//   xb:0  wq:16M  wk:24M  wv:32M  wo:40M  q:48M  k:64M  vT:80M  ao:96M (112M)
// ---------------------------------------------------------------------------
extern "C" void kernel_launch(void* const* d_in, const int* in_sizes, int n_in,
                              void* d_out, int out_size, void* d_ws, size_t ws_size,
                              hipStream_t stream) {
    const float* x  = (const float*)d_in[0];
    // d_in[1] is the causal mask; causality is applied analytically.
    const float* wq = (const float*)d_in[2];
    const float* wk = (const float*)d_in[3];
    const float* wv = (const float*)d_in[4];
    const float* wo = (const float*)d_in[5];
    float* out = (float*)d_out;

    char* ws = (char*)d_ws;
    bf16_t* xb  = (bf16_t*)(ws + 0);
    bf16_t* wqb = (bf16_t*)(ws + (size_t)16 * 1024 * 1024);
    bf16_t* wkb = (bf16_t*)(ws + (size_t)24 * 1024 * 1024);
    bf16_t* wvb = (bf16_t*)(ws + (size_t)32 * 1024 * 1024);
    bf16_t* wob = (bf16_t*)(ws + (size_t)40 * 1024 * 1024);
    bf16_t* qb  = (bf16_t*)(ws + (size_t)48 * 1024 * 1024);
    bf16_t* kb  = (bf16_t*)(ws + (size_t)64 * 1024 * 1024);
    bf16_t* vtb = (bf16_t*)(ws + (size_t)80 * 1024 * 1024);
    bf16_t* aob = (bf16_t*)(ws + (size_t)96 * 1024 * 1024);

    const int nX = Mrows * Dc;   // 8M
    const int nW = Dc * Dc;      // 4M

    cvt_f32_bf16<<<4096, 256, 0, stream>>>(x,  xb,  nX);
    cvt_f32_bf16<<<4096, 256, 0, stream>>>(wq, wqb, nW);
    cvt_f32_bf16<<<4096, 256, 0, stream>>>(wk, wkb, nW);
    cvt_f32_bf16<<<4096, 256, 0, stream>>>(wv, wvb, nW);
    cvt_f32_bf16<<<4096, 256, 0, stream>>>(wo, wob, nW);

    dim3 gg(Dc / 64, Mrows / 128);   // 32 x 32 blocks
    gemm_xwt<0><<<gg, 256, 0, stream>>>(xb, wqb, qb,  nullptr);
    gemm_xwt<0><<<gg, 256, 0, stream>>>(xb, wkb, kb,  nullptr);
    gemm_xwt<1><<<gg, 256, 0, stream>>>(xb, wvb, vtb, nullptr);

    flash_attn<<<Bc * Hc * (Sc / 128), 256, 0, stream>>>(qb, kb, vtb, aob);

    gemm_xwt<2><<<gg, 256, 0, stream>>>(aob, wob, nullptr, out);
}
